// SingleStreamBlock_15710990368997
// MI455X (gfx1250) — matrix operand order repaired and checked by reading, not compiled, not run on hardware
//
#include <hip/hip_runtime.h>
#include <hip/hip_bf16.h>
#include <math.h>

typedef __attribute__((ext_vector_type(16))) __bf16 v16bf;
typedef __attribute__((ext_vector_type(8)))  float  v8f;
typedef __attribute__((ext_vector_type(4)))  unsigned int u32x4;
typedef __attribute__((ext_vector_type(8)))  unsigned int u32x8;

union Frag { v16bf v; unsigned int u[8]; };

// Native bf16 converts (RNE) -> v_cvt_(pk_)bf16_f32 on gfx1250
__device__ __forceinline__ unsigned short f2bf(float f) {
  union { __bf16 b; unsigned short s; } cv;
  cv.b = (__bf16)f;
  return cv.s;
}
__device__ __forceinline__ unsigned int pack_bf16(float lo, float hi) {
  union { __bf16 b[2]; unsigned int u; } cv;
  cv.b[0] = (__bf16)lo;
  cv.b[1] = (__bf16)hi;
  return cv.u;
}

// ISA 16-bit A-matrix 16x32 layout: VGPR j holds K pair; lane-half lh selects K offset.
__device__ __forceinline__ int apos(int j, int lh) {
  return (j < 4) ? (lh * 8 + 2 * j) : (16 + lh * 8 + 2 * (j - 4));
}

// ---- Tensor Data Mover: 2D bf16 tile Global -> LDS (ISA 08_async_tensor §8) ----
// D# group0: count=1 | lds_addr | global_addr[56:0] | type=2
// D# group1: data_size=2B | tensor_dim0/1 | tile_dim0/1 | tensor_dim0_stride
__device__ __forceinline__ void tdm_load_2d_bf16(
    const void* gptr, unsigned int lds_off,
    unsigned int tile_d0, unsigned int tile_d1,
    unsigned int tensor_d0, unsigned int tensor_d1,
    unsigned int stride_d0 /* elements */) {
  unsigned long long ga = (unsigned long long)(size_t)gptr;
  u32x4 g0;
  g0[0] = 1u;                                             // count=1 (valid), user mode
  g0[1] = lds_off;                                        // lds_addr (bytes)
  g0[2] = (unsigned int)(ga & 0xFFFFFFFFull);             // global_addr[31:0]
  g0[3] = (unsigned int)((ga >> 32) & 0x1FFFFFFull) | (2u << 30);  // addr[56:32], type=2
  u32x8 g1;
  g1[0] = 1u << 16;                                       // data_size = 1 -> 2 bytes
  g1[1] = (tensor_d0 & 0xFFFFu) << 16;                    // tensor_dim0[15:0]
  g1[2] = (tensor_d0 >> 16) | ((tensor_d1 & 0xFFFFu) << 16);
  g1[3] = (tensor_d1 >> 16) | (tile_d0 << 16);            // tile_dim0
  g1[4] = tile_d1;                                        // tile_dim1 (tile_dim2=0)
  g1[5] = stride_d0;                                      // tensor_dim0_stride[31:0]
  g1[6] = 0u;
  g1[7] = 0u;
  asm volatile("tensor_load_to_lds %0, %1" :: "s"(g0), "s"(g1) : "memory");
}

// ---------------------------------------------------------------- silu(vec)
__global__ void silu_k(const float* __restrict__ vec, float* __restrict__ sv) {
  int i = blockIdx.x * 256 + threadIdx.x;
  if (i < 3072) { float v = vec[i]; sv[i] = v / (1.0f + __expf(-v)); }
}

// ------------------------------------------- mod = silu(vec) @ mod_w + mod_b
__global__ __launch_bounds__(256) void mod_gemv(const float* __restrict__ sv,
                                                const float* __restrict__ mw,
                                                const float* __restrict__ mb,
                                                float* __restrict__ mod) {
  int c = blockIdx.x * 256 + threadIdx.x;  // < 9216
  float acc = mb[c];
  #pragma unroll 4
  for (int r = 0; r < 3072; r++) acc += sv[r] * mw[(size_t)r * 9216 + c];
  mod[c] = acc;
}

// -------------------------- x_mod = (1+scale)*LN(x) + shift, stored as bf16
__global__ __launch_bounds__(256) void ln_mod(const float* __restrict__ x,
                                              const float* __restrict__ mod,
                                              unsigned short* __restrict__ xmod) {
  int l = blockIdx.x, t = threadIdx.x;
  const float* xr = x + (size_t)l * 3072;
  float vals[12]; float s = 0.f, ss = 0.f;
  #pragma unroll
  for (int i = 0; i < 12; i++) { float v = xr[t + i * 256]; vals[i] = v; s += v; ss += v * v; }
  #pragma unroll
  for (int m = 16; m >= 1; m >>= 1) { s += __shfl_xor(s, m, 32); ss += __shfl_xor(ss, m, 32); }
  __shared__ float red[16];
  int w = t >> 5, lane = t & 31;
  if (lane == 0) { red[w] = s; red[8 + w] = ss; }
  __syncthreads();
  if (t == 0) {
    float S = 0.f, SS = 0.f;
    for (int i = 0; i < 8; i++) { S += red[i]; SS += red[8 + i]; }
    float mu = S / 3072.f;
    float var = SS / 3072.f - mu * mu;
    red[0] = mu; red[1] = rsqrtf(var + 1e-6f);
  }
  __syncthreads();
  float mu = red[0], rstd = red[1];
  #pragma unroll
  for (int i = 0; i < 12; i++) {
    int c = t + i * 256;
    float xm = (1.f + mod[3072 + c]) * ((vals[i] - mu) * rstd) + mod[c];
    xmod[(size_t)l * 3072 + c] = f2bf(xm);
  }
}

// ------------------------------------------------------------- GEMM template
// C[M=2048, N] = A_bf16[M,K] @ Bw_f32[K,N] (+bias, fused epilogue)
// 128x128 C tile / block, 8 waves x (32x64). A tile via TDM, B reg-pipelined.
template <int MODE>
__global__ __launch_bounds__(256) void gemm_bf16w(
    const unsigned short* __restrict__ A, int K,
    const float* __restrict__ Bw, int Nt,
    const float* __restrict__ bias,
    float* __restrict__ out_f,            // MODE0: qkv[2048,9216]  MODE1: out[2048,3072]
    unsigned short* __restrict__ out_bf,  // MODE0: concat[2048,15360]
    const float* __restrict__ xres,       // MODE1: x
    const float* __restrict__ modv)       // MODE1: mod (gate at +6144)
{
  __shared__ __align__(16) unsigned short As[2][128 * 32];  // [m][k]
  __shared__ __align__(16) unsigned short Bs[2][128 * 32];  // [n][k] (transposed)
  const int tid = threadIdx.x;
  const int lane = tid & 31, w = tid >> 5;
  const int ln = lane & 15, lh = lane >> 4;
  const int m0 = blockIdx.y * 128, n0 = blockIdx.x * 128;
  const int waveM = (w & 3) * 32, waveN = (w >> 2) * 64;

  v8f acc[2][4];
  #pragma unroll
  for (int i = 0; i < 2; i++)
    #pragma unroll
    for (int j = 0; j < 4; j++) acc[i][j] = (v8f){0, 0, 0, 0, 0, 0, 0, 0};

  // B staging: 2 consecutive k-rows x 8 cols per thread, packed u32 stores
  const int kr2 = (tid & 15) * 2;
  const int cg = (tid >> 4) * 8;
  float4 b0a, b0b, b1a, b1b;

  auto loadB = [&](int kk) {
    const float* p0 = Bw + (size_t)(kk + kr2) * Nt + n0 + cg;
    b0a = *(const float4*)p0;
    b0b = *(const float4*)(p0 + 4);
    b1a = *(const float4*)(p0 + Nt);
    b1b = *(const float4*)(p0 + Nt + 4);
  };
  auto storeB = [&](int buf) {
    float r0[8] = {b0a.x, b0a.y, b0a.z, b0a.w, b0b.x, b0b.y, b0b.z, b0b.w};
    float r1[8] = {b1a.x, b1a.y, b1a.z, b1a.w, b1b.x, b1b.y, b1b.z, b1b.w};
    #pragma unroll
    for (int j = 0; j < 8; j++)
      *(unsigned int*)&Bs[buf][(cg + j) * 32 + kr2] = pack_bf16(r0[j], r1[j]);
  };

  // prologue: TDM A tile 0, B regs tile 0
  if (w == 0)
    tdm_load_2d_bf16(A + (size_t)m0 * K, (unsigned int)(size_t)&As[0][0],
                     32, 128, 32, 128, (unsigned int)K);
  loadB(0);

  int cur = 0;
  for (int kk = 0; kk < K; kk += 32, cur ^= 1) {
    storeB(cur);
    __builtin_amdgcn_s_wait_tensorcnt(0);
    __syncthreads();
    if (kk + 32 < K) {
      if (w == 0)
        tdm_load_2d_bf16(A + (size_t)m0 * K + kk + 32,
                         (unsigned int)(size_t)&As[cur ^ 1][0],
                         32, 128, 32, 128, (unsigned int)K);
      loadB(kk + 32);
    }

    Frag af[2];
    #pragma unroll
    for (int ti = 0; ti < 2; ti++)
      #pragma unroll
      for (int j = 0; j < 8; j++)
        af[ti].u[j] = *(const unsigned int*)&As[cur][(waveM + 16 * ti + ln) * 32 + apos(j, lh)];
    Frag bf[4];
    #pragma unroll
    for (int tj = 0; tj < 4; tj++)
      #pragma unroll
      for (int j = 0; j < 8; j++)
        bf[tj].u[j] = *(const unsigned int*)&Bs[cur][(waveN + 16 * tj + ln) * 32 + apos(j, lh)];

    #pragma unroll
    for (int ti = 0; ti < 2; ti++)
      #pragma unroll
      for (int tj = 0; tj < 4; tj++)
        acc[ti][tj] = __builtin_amdgcn_wmma_f32_16x16x32_bf16(
            false, af[ti].v, false, bf[tj].v, (short)0, acc[ti][tj], false, false);
  }

  // epilogue (C layout: VGPR r -> m = r + 8*lh, n = ln)
  #pragma unroll
  for (int ti = 0; ti < 2; ti++)
    #pragma unroll
    for (int tj = 0; tj < 4; tj++)
      #pragma unroll
      for (int r = 0; r < 8; r++) {
        int m = m0 + waveM + 16 * ti + r + 8 * lh;
        int n = n0 + waveN + 16 * tj + ln;
        float val = acc[ti][tj][r] + bias[n];
        if (MODE == 0) {
          if (n < 9216) {
            out_f[(size_t)m * 9216 + n] = val;
          } else {
            float xx = val;
            float gl = 0.5f * xx * (1.0f + tanhf(0.7978845608028654f * (xx + 0.044715f * xx * xx * xx)));
            out_bf[(size_t)m * 15360 + 3072 + (n - 9216)] = f2bf(gl);
          }
        } else {
          out_f[(size_t)m * 3072 + n] =
              xres[(size_t)m * 3072 + n] + modv[6144 + n] * val;
        }
      }
}

// ------------------------- q/k RMSNorm + RoPE + scale; v passthrough; -> bf16
__global__ __launch_bounds__(32) void qkv_prep(
    const float* __restrict__ qkv, const float* __restrict__ pe,
    const float* __restrict__ qnw, const float* __restrict__ knw,
    unsigned short* __restrict__ qb, unsigned short* __restrict__ kb,
    unsigned short* __restrict__ vb) {
  int l = blockIdx.x, h = blockIdx.y, t = threadIdx.x;  // 32 threads, 2 pairs each
  const float* base = qkv + (size_t)l * 9216 + h * 128;
  int d0 = 2 * t, d1 = 2 * t + 1, d2 = 64 + 2 * t, d3 = 65 + 2 * t;
  float q0 = base[d0], q1 = base[d1], q2 = base[d2], q3 = base[d3];
  float k0 = base[3072 + d0], k1 = base[3072 + d1], k2 = base[3072 + d2], k3 = base[3072 + d3];
  float v0 = base[6144 + d0], v1 = base[6144 + d1], v2 = base[6144 + d2], v3 = base[6144 + d3];
  float sq = q0 * q0 + q1 * q1 + q2 * q2 + q3 * q3;
  float sk = k0 * k0 + k1 * k1 + k2 * k2 + k3 * k3;
  #pragma unroll
  for (int m = 16; m >= 1; m >>= 1) { sq += __shfl_xor(sq, m, 32); sk += __shfl_xor(sk, m, 32); }
  float rq = rsqrtf(sq / 128.f + 1e-6f), rk = rsqrtf(sk / 128.f + 1e-6f);
  q0 *= rq * qnw[d0]; q1 *= rq * qnw[d1]; q2 *= rq * qnw[d2]; q3 *= rq * qnw[d3];
  k0 *= rk * knw[d0]; k1 *= rk * knw[d1]; k2 *= rk * knw[d2]; k3 *= rk * knw[d3];
  const float* per = pe + (size_t)l * 256;  // [64 pairs][2][2]
  int p0 = t * 4, p1 = (t + 32) * 4;
  float nq0 = per[p0 + 0] * q0 + per[p0 + 1] * q1;
  float nq1 = per[p0 + 2] * q0 + per[p0 + 3] * q1;
  float nq2 = per[p1 + 0] * q2 + per[p1 + 1] * q3;
  float nq3 = per[p1 + 2] * q2 + per[p1 + 3] * q3;
  float nk0 = per[p0 + 0] * k0 + per[p0 + 1] * k1;
  float nk1 = per[p0 + 2] * k0 + per[p0 + 3] * k1;
  float nk2 = per[p1 + 0] * k2 + per[p1 + 1] * k3;
  float nk3 = per[p1 + 2] * k2 + per[p1 + 3] * k3;
  const float sc = 0.08838834764831845f;  // 128^-0.5 folded into q
  size_t ob = ((size_t)h * 2048 + l) * 128;
  *(unsigned int*)&qb[ob + d0] = pack_bf16(nq0 * sc, nq1 * sc);
  *(unsigned int*)&qb[ob + d2] = pack_bf16(nq2 * sc, nq3 * sc);
  *(unsigned int*)&kb[ob + d0] = pack_bf16(nk0, nk1);
  *(unsigned int*)&kb[ob + d2] = pack_bf16(nk2, nk3);
  *(unsigned int*)&vb[ob + d0] = pack_bf16(v0, v1);
  *(unsigned int*)&vb[ob + d2] = pack_bf16(v2, v3);
}

// ------------------------------------------ flash attention, bf16 WMMA, f32 acc
__global__ __launch_bounds__(256) void attn_kernel(
    const unsigned short* __restrict__ qg, const unsigned short* __restrict__ kg,
    const unsigned short* __restrict__ vg, unsigned short* __restrict__ concat) {
  __shared__ __align__(16) unsigned short Ks[64 * 128];    // [key][dim] == Bt for QK^T
  __shared__ __align__(16) unsigned short Vs[128 * 64];    // [dim][key] == Bt for P@V
  __shared__ __align__(16) unsigned short Ps[8][16 * 64];  // per-wave P slab
  const int h = blockIdx.x, q0 = blockIdx.y * 128;
  const int tid = threadIdx.x, w = tid >> 5, lane = tid & 31;
  const int ln = lane & 15, lh = lane >> 4;
  const size_t hb = (size_t)h * 2048 * 128;
  const unsigned short* Q = qg + hb;
  const unsigned short* Kp = kg + hb;
  const unsigned short* Vp = vg + hb;

  // preload Q fragments (A layout), 16 rows per wave, 4 dim-chunks of 32
  Frag qf[4];
  const int qm = q0 + w * 16 + ln;
  #pragma unroll
  for (int dc = 0; dc < 4; dc++)
    #pragma unroll
    for (int j = 0; j < 8; j++)
      qf[dc].u[j] = *(const unsigned int*)&Q[(size_t)qm * 128 + dc * 32 + apos(j, lh)];

  float mi[8], li[8];
  v8f o[8];
  #pragma unroll
  for (int r = 0; r < 8; r++) { mi[r] = -3.0e38f; li[r] = 0.f; }
  #pragma unroll
  for (int tn = 0; tn < 8; tn++) o[tn] = (v8f){0, 0, 0, 0, 0, 0, 0, 0};

  const int key2 = (tid & 31) * 2, db = (tid >> 5) * 16;  // V transpose staging

  for (int kb0 = 0; kb0 < 2048; kb0 += 64) {
    // K tile [64 keys][128 dims] via Tensor Data Mover
    if (w == 0)
      tdm_load_2d_bf16(Kp + (size_t)kb0 * 128, (unsigned int)(size_t)Ks,
                       128, 64, 128, 64, 128);
    // V tile transposed -> Vs[dim][key], packed u32 stores (overlaps the DMA)
    unsigned int a0[8], a1[8];
    #pragma unroll
    for (int j = 0; j < 8; j++) {
      a0[j] = *(const unsigned int*)&Vp[(size_t)(kb0 + key2) * 128 + db + 2 * j];
      a1[j] = *(const unsigned int*)&Vp[(size_t)(kb0 + key2 + 1) * 128 + db + 2 * j];
    }
    #pragma unroll
    for (int j = 0; j < 8; j++) {
      *(unsigned int*)&Vs[(db + 2 * j) * 64 + key2] = (a0[j] & 0xFFFFu) | (a1[j] << 16);
      *(unsigned int*)&Vs[(db + 2 * j + 1) * 64 + key2] = (a0[j] >> 16) | (a1[j] & 0xFFFF0000u);
    }
    __builtin_amdgcn_s_wait_tensorcnt(0);
    __syncthreads();

    // S = Q K^T  (16 x 64 per wave)
    v8f s[4];
    #pragma unroll
    for (int tj = 0; tj < 4; tj++) s[tj] = (v8f){0, 0, 0, 0, 0, 0, 0, 0};
    #pragma unroll
    for (int tj = 0; tj < 4; tj++)
      #pragma unroll
      for (int dc = 0; dc < 4; dc++) {
        Frag bfr;
        #pragma unroll
        for (int j = 0; j < 8; j++)
          bfr.u[j] = *(const unsigned int*)&Ks[(tj * 16 + ln) * 128 + dc * 32 + apos(j, lh)];
        s[tj] = __builtin_amdgcn_wmma_f32_16x16x32_bf16(
            false, qf[dc].v, false, bfr.v, (short)0, s[tj], false, false);
      }

    // online softmax (row m = r + 8*lh lives in a 16-lane group)
    #pragma unroll
    for (int r = 0; r < 8; r++) {
      float mx = fmaxf(fmaxf(s[0][r], s[1][r]), fmaxf(s[2][r], s[3][r]));
      #pragma unroll
      for (int m = 8; m >= 1; m >>= 1) mx = fmaxf(mx, __shfl_xor(mx, m, 32));
      float mnew = fmaxf(mi[r], mx);
      float alpha = __expf(mi[r] - mnew);
      mi[r] = mnew;
      float rs = 0.f;
      #pragma unroll
      for (int tj = 0; tj < 4; tj++) {
        float pv = __expf(s[tj][r] - mnew);
        s[tj][r] = pv;
        rs += pv;
      }
      #pragma unroll
      for (int m = 8; m >= 1; m >>= 1) rs += __shfl_xor(rs, m, 32);
      li[r] = li[r] * alpha + rs;
      #pragma unroll
      for (int tn = 0; tn < 8; tn++) o[tn][r] *= alpha;
      #pragma unroll
      for (int tj = 0; tj < 4; tj++)
        Ps[w][(r + 8 * lh) * 64 + tj * 16 + ln] = f2bf(s[tj][r]);
    }

    // O += P @ V  (wave-private LDS slab, DS in-order per wave)
    #pragma unroll
    for (int kc = 0; kc < 2; kc++) {
      Frag pf;
      #pragma unroll
      for (int j = 0; j < 8; j++)
        pf.u[j] = *(const unsigned int*)&Ps[w][ln * 64 + kc * 32 + apos(j, lh)];
      #pragma unroll
      for (int tn = 0; tn < 8; tn++) {
        Frag bfr;
        #pragma unroll
        for (int j = 0; j < 8; j++)
          bfr.u[j] = *(const unsigned int*)&Vs[(tn * 16 + ln) * 64 + kc * 32 + apos(j, lh)];
        o[tn] = __builtin_amdgcn_wmma_f32_16x16x32_bf16(
            false, pf.v, false, bfr.v, (short)0, o[tn], false, false);
      }
    }
    __syncthreads();
  }

  // normalize and write attn -> concat cols [0,3072)
  float inv[8];
  #pragma unroll
  for (int r = 0; r < 8; r++) inv[r] = 1.0f / li[r];
  #pragma unroll
  for (int tn = 0; tn < 8; tn++)
    #pragma unroll
    for (int r = 0; r < 8; r++) {
      int m = q0 + w * 16 + r + 8 * lh;
      int col = h * 128 + tn * 16 + ln;
      concat[(size_t)m * 15360 + col] = f2bf(o[tn][r] * inv[r]);
    }
}

// ---------------------------------------------------------------------------
extern "C" void kernel_launch(void* const* d_in, const int* in_sizes, int n_in,
                              void* d_out, int out_size, void* d_ws, size_t ws_size,
                              hipStream_t stream) {
  const float* x     = (const float*)d_in[0];
  const float* vec   = (const float*)d_in[1];
  const float* pe    = (const float*)d_in[2];
  const float* w1    = (const float*)d_in[3];
  const float* b1    = (const float*)d_in[4];
  const float* w2    = (const float*)d_in[5];
  const float* b2    = (const float*)d_in[6];
  const float* mod_w = (const float*)d_in[7];
  const float* mod_b = (const float*)d_in[8];
  const float* qnw   = (const float*)d_in[9];
  const float* knw   = (const float*)d_in[10];
  float* out = (float*)d_out;

  char* ws = (char*)d_ws;
  size_t off = 0;
  auto wsa = [&](size_t bytes) {
    void* p = ws + off;
    off = (off + bytes + 255) & ~(size_t)255;
    return p;
  };
  float* sv             = (float*)wsa(3072 * 4);
  float* mod            = (float*)wsa(9216 * 4);
  unsigned short* xmod  = (unsigned short*)wsa((size_t)2048 * 3072 * 2);
  float* qkv            = (float*)wsa((size_t)2048 * 9216 * 4);
  unsigned short* cat   = (unsigned short*)wsa((size_t)2048 * 15360 * 2);
  unsigned short* qb    = (unsigned short*)wsa((size_t)24 * 2048 * 128 * 2);
  unsigned short* kb    = (unsigned short*)wsa((size_t)24 * 2048 * 128 * 2);
  unsigned short* vb    = (unsigned short*)wsa((size_t)24 * 2048 * 128 * 2);

  silu_k<<<12, 256, 0, stream>>>(vec, sv);
  mod_gemv<<<36, 256, 0, stream>>>(sv, mod_w, mod_b, mod);
  ln_mod<<<2048, 256, 0, stream>>>(x, mod, xmod);
  gemm_bf16w<0><<<dim3(168, 16), 256, 0, stream>>>(xmod, 3072, w1, 21504, b1,
                                                   qkv, cat, nullptr, nullptr);
  qkv_prep<<<dim3(2048, 24), 32, 0, stream>>>(qkv, pe, qnw, knw, qb, kb, vb);
  attn_kernel<<<dim3(24, 16), 256, 0, stream>>>(qb, kb, vb, cat);
  gemm_bf16w<1><<<dim3(24, 16), 256, 0, stream>>>(cat, 15360, w2, 3072, b2,
                                                  out, nullptr, x, mod);
}
